// leaky_integrator_conductance_94489280741
// MI455X (gfx1250) — matrix operand order repaired and checked by reading, not compiled, not run on hardware
//
#include <hip/hip_runtime.h>
#include <hip/hip_bf16.h>

#define Bsz 16
#define Nn 300
#define Tt 4096
#define NPAD 304     // i rows padded to 16
#define KPAD 320     // j (K dim) padded to 32
#define KSTR 328     // LDS row stride in shorts (164 dwords -> conflict-free b128)
#define TTILE 64     // t columns per block
#define NTHR 128     // 4 waves

typedef __attribute__((ext_vector_type(16))) __bf16 v16bf;
typedef __attribute__((ext_vector_type(8)))  float  v8f;

__device__ __forceinline__ unsigned int f2bf(float f) {
    union { float f; unsigned int u; } c; c.f = f;
    unsigned int u = c.u;
    return (u + 0x7FFFu + ((u >> 16) & 1u)) >> 16;   // RNE bf16 in low 16 bits
}

// ---------------------------------------------------------------------------
// Kernel 1: per-row scalars + bf16 W_e panel (zero padded to [NPAD][KPAD])
// ---------------------------------------------------------------------------
__global__ __launch_bounds__(64) void lic_prep(
    const float* __restrict__ sc, const float* __restrict__ se,
    const float* __restrict__ mc, const float* __restrict__ me,
    const float* __restrict__ E,  const float* __restrict__ tau,
    float* __restrict__ Av, float* __restrict__ Sv, float* __restrict__ Sev,
    float* __restrict__ dtau, unsigned short* __restrict__ wbf)
{
    const int i   = blockIdx.x;      // 0..NPAD-1
    const int tid = threadIdx.x;     // 0..63
    __shared__ float red[3][64];

    float a = 0.f, s = 0.f, ss = 0.f;
    for (int j = tid; j < KPAD; j += 64) {
        float we = 0.f;
        if (i < Nn && j < Nn) {
            const float wc = sc[i*Nn + j] * mc[i*Nn + j] * 0.01f;
            a += wc * E[i*Nn + j];
            s += wc;
            we = se[i*Nn + j] * (me[i*Nn + j] + me[j*Nn + i]) * 0.01f;
            ss += we;
        }
        wbf[i*KPAD + j] = (unsigned short)f2bf(we);
    }
    red[0][tid] = a; red[1][tid] = s; red[2][tid] = ss;
    __syncthreads();
    for (int off = 32; off > 0; off >>= 1) {
        if (tid < off) {
            red[0][tid] += red[0][tid + off];
            red[1][tid] += red[1][tid + off];
            red[2][tid] += red[2][tid + off];
        }
        __syncthreads();
    }
    if (tid == 0) {
        Av[i] = red[0][0]; Sv[i] = red[1][0]; Sev[i] = red[2][0];
        dtau[i] = (i < Nn) ? (0.01f / tau[i]) : 0.f;
    }
}

// ---------------------------------------------------------------------------
// Kernel 2: fused WMMA GEMM (elec) + chem + leaky-integrator epilogue.
// Block: 128 threads (4 waves). Tile: 16 i-rows x 64 t-cols.
// Full K panel staged once in LDS; single barrier; 10 unrolled WMMAs/wave.
// ---------------------------------------------------------------------------
__global__ __launch_bounds__(NTHR) void lic_main(
    const float* __restrict__ input, const float* __restrict__ hidden,
    const float* __restrict__ Av, const float* __restrict__ Sv,
    const float* __restrict__ Sev, const float* __restrict__ dtau,
    const float* __restrict__ bias, const unsigned short* __restrict__ wbf,
    float* __restrict__ out)
{
    const int t0  = blockIdx.x * TTILE;
    const int i0  = blockIdx.y * 16;
    const int b   = blockIdx.z;
    const int tid = threadIdx.x;
    const int wave = tid >> 5;        // 0..3 -> t sub-tile
    const int lane = tid & 31;
    const int l15  = lane & 15;
    const int hrow = lane >> 4;       // 0 or 1 (half-wave)

    __shared__ unsigned short Wl[16 * KSTR];      // 10.5 KB, W_e rows (bf16)
    __shared__ unsigned short Xl[TTILE * KSTR];   // 42 KB, x panel [t][k] (bf16)
    __shared__ float sA[16], sS[16], sSe[16], sDt[16], sB[16];

    // --- stage epilogue scalars (16 rows) ---
    if (tid < 16) {
        const int i = i0 + tid;       // <= 303 < NPAD, always in-bounds in ws
        sA[tid]  = Av[i];
        sS[tid]  = Sv[i];
        sSe[tid] = Sev[i];
        sDt[tid] = dtau[i];
        sB[tid]  = (i < Nn) ? bias[i] : 0.f;
    }

    // --- stage 16 W rows (stride KPAD -> padded KSTR): 2560 dwords ---
    {
        const unsigned int* wsrc = (const unsigned int*)wbf;
        unsigned int* wdst = (unsigned int*)Wl;
        #pragma unroll
        for (int r = 0; r < 20; ++r) {
            const int idx = r * NTHR + tid;          // 0..2559
            const int row = idx / 160;
            const int col = idx - row * 160;
            wdst[row * (KSTR / 2) + col] = wsrc[(size_t)(i0 + row) * 160 + col];
        }
    }

    // --- stage full time-shifted x panel: x[b, j, t0+t] -> Xl[t][j] bf16 ---
    // Unconditional clamped loads + select (no exec divergence); two bf16
    // packed per ds_store_b32. 64 t x 160 dword-columns = 10240 dwords.
    const float* hb = hidden + (size_t)b * Nn * Tt;
    {
        unsigned int* xdst = (unsigned int*)Xl;
        #pragma unroll 4
        for (int rep = 0; rep < 80; ++rep) {
            const int idx  = rep * NTHR + tid;
            const int tloc = idx & (TTILE - 1);
            const int kp   = idx >> 6;               // 0..159
            const int jg   = 2 * kp;
            const int tg   = t0 + tloc;
            const int ts   = (tg > 0) ? (tg - 1) : 0;        // clamped shift
            const int j0c  = (jg     < Nn) ? jg       : 0;   // clamped rows
            const int j1c  = (jg + 1 < Nn) ? (jg + 1) : 0;
            float v0 = hb[(size_t)j0c * Tt + ts];
            float v1 = hb[(size_t)j1c * Tt + ts];
            v0 = (tg > 0 && jg     < Nn) ? v0 : 0.f;
            v1 = (tg > 0 && jg + 1 < Nn) ? v1 : 0.f;
            xdst[tloc * (KSTR / 2) + kp] = f2bf(v0) | (f2bf(v1) << 16);
        }
    }

    __syncthreads();

    // --- 10 back-to-back WMMAs over K = 320 ---
    const int tl = wave * 16 + l15;                  // t column (local)
    v8f acc = {};
    const unsigned short* wr0 = Wl + l15 * KSTR + (hrow ? 8 : 0);
    const unsigned short* xr0 = Xl + tl  * KSTR + (hrow ? 8 : 0);
    #pragma unroll
    for (int kc = 0; kc < KPAD; kc += 32) {
        union { unsigned int u[8]; v16bf v; } af, bfr;
        const unsigned short* wr = wr0 + kc;
        const unsigned short* xr = xr0 + kc;
        #pragma unroll
        for (int q = 0; q < 4; ++q) {
            af.u[q]     = *(const unsigned int*)(wr + 2 * q);        // K pairs 0..7
            af.u[q + 4] = *(const unsigned int*)(wr + 16 + 2 * q);   // K pairs 16..23
        }
        #pragma unroll
        for (int q = 0; q < 8; ++q)
            bfr.u[q] = *(const unsigned int*)(xr + 2 * q + (q >= 4 ? 8 : 0)); // K pairs 0..7 / 16..23
        acc = __builtin_amdgcn_wmma_f32_16x16x32_bf16(
            /*neg_a=*/false, af.v, /*neg_b=*/false, bfr.v,
            /*c_mod=*/(short)0, acc, /*reuse_a=*/false, /*reuse_b=*/false);
    }

    // --- epilogue: C/D layout -> VGPR r holds row i0+r (+8 for lanes 16-31),
    // column t0 + tl ---
    const int tg = t0 + tl;
    const float* inb = input + (size_t)b * Nn * Tt;
    const size_t plane = (size_t)Bsz * Nn * Tt;
    #pragma unroll
    for (int r = 0; r < 8; ++r) {
        const int rl = r + (hrow ? 8 : 0);
        const int i  = i0 + rl;
        if (i < Nn) {
            const float x    = (tg > 0) ? hb[(size_t)i * Tt + (tg - 1)] : 0.f;
            const float elec = acc[r] - sSe[rl] * x;
            const float rlu  = x > 0.f ? x : 0.f;
            const float chem = (sA[rl] - x * sS[rl]) * rlu;
            const float inp  = inb[(size_t)i * Tt + tg];
            const float mu   = sDt[rl] * (chem + elec + sB[rl] + inp - x) + x;
            const size_t o   = ((size_t)b * Nn + i) * Tt + tg;
            out[o]             = mu;
            out[plane + o]     = chem;
            out[2 * plane + o] = elec;
        }
    }
}

extern "C" void kernel_launch(void* const* d_in, const int* in_sizes, int n_in,
                              void* d_out, int out_size, void* d_ws, size_t ws_size,
                              hipStream_t stream) {
    const float* input  = (const float*)d_in[0];
    const float* hidden = (const float*)d_in[1];
    const float* sc     = (const float*)d_in[2];
    const float* se     = (const float*)d_in[3];
    const float* mc     = (const float*)d_in[4];
    const float* me     = (const float*)d_in[5];
    const float* E      = (const float*)d_in[6];   // [1,N,N] -> N*N
    const float* bias   = (const float*)d_in[7];
    const float* tau    = (const float*)d_in[8];
    float* out = (float*)d_out;

    // Workspace: 4 f32 vectors of NPAD, then bf16 W_e [NPAD][KPAD].
    float* wsf = (float*)d_ws;
    float* Av   = wsf;
    float* Sv   = wsf + NPAD;
    float* Sev  = wsf + 2 * NPAD;
    float* dtau = wsf + 3 * NPAD;
    unsigned short* wbf = (unsigned short*)(wsf + 4 * NPAD);

    lic_prep<<<NPAD, 64, 0, stream>>>(sc, se, mc, me, E, tau, Av, Sv, Sev, dtau, wbf);

    dim3 grid(Tt / TTILE, NPAD / 16, Bsz);   // 64 x 19 x 16
    lic_main<<<grid, NTHR, 0, stream>>>(input, hidden, Av, Sv, Sev, dtau, bias, wbf, out);
}
